// ProjAttn_5506148074203
// MI455X (gfx1250) — matrix operand order repaired and checked by reading, not compiled, not run on hardware
//
#include <hip/hip_runtime.h>
#include <hip/hip_bf16.h>

// Problem constants (fixed by the reference)
#define V_   5
#define LQ_  4096
#define C_   256
#define H_   128
#define W_   240
#define HW_  (H_ * W_)
#define NH_  8
#define P_   4
#define DH_  32

typedef __attribute__((ext_vector_type(16))) _Float16 v16h;
typedef __attribute__((ext_vector_type(8)))  float    v8f;

union FragH  { uint4 u[2]; v16h v; };
union PackU  { _Float16 h[2]; unsigned u; };
union Pack2U { _Float16 h[4]; uint2 u2; };

// ---------------------------------------------------------------------------
// Templated K=256 GEMM: C[m,n] = sum_k A[m,k]*B[k,n] (+ bias, + optional
// rank-3 camera-ray term). fp32 in/out, f16 WMMA 16x16x32 with fp32 acc.
//   NT    : number of 16-wide N tiles per block (BN = 16*NT)
//   ACOL  : A stored column-major (A[m,k] = A[k*lda + m])  -- src0 case
//   CAM   : fuse value += cam[m,0..2] @ Wcam[0..2, n] epilogue
// Block = 256 threads = 8 waves; BM = 128 (16 rows per wave); BK = 32.
// Double-buffered LDS; per-tile staging is load-phase -> store-phase so all
// global_load_b128 are in flight together, overlapped with WMMA of the
// previous tile (one barrier per K-tile).
// ---------------------------------------------------------------------------
template<int NT, bool ACOL, bool CAM>
__global__ __launch_bounds__(256)
void gemm_k256(const float* __restrict__ A, long long aBatch, int lda,
               const float* __restrict__ B, int ldb,
               const float* __restrict__ bias,
               const float* __restrict__ cam, long long camBatch,
               const float* __restrict__ Wcam,
               float* __restrict__ Cp, long long cBatch, int ldc)
{
    constexpr int BM = 128, BK = 32, BN = NT * 16;
    constexpr int KT = 256 / BK;                 // 8 K-tiles
    // stride 40 halves = 80B: 16B-aligned fragment reads, mild bank conflicts
    __shared__ _Float16 Al[2][BM][40];
    __shared__ _Float16 Bl[2][BN][40];

    const int tid  = threadIdx.x;
    const int wave = tid >> 5;
    const int lane = tid & 31;
    const int l16  = lane & 15;
    const int lhi  = lane >> 4;
    const int mBase = blockIdx.x * BM;
    const int nBase = blockIdx.y * BN;
    const long long batch = blockIdx.z;

    const float* Ab = A + batch * aBatch;
    float*       Cb = Cp + batch * cBatch;

    // ---- per-thread staging geometry (hoisted: no muls in the hot loop) ----
    const float* aP[4];
    int aM[2], aKp[2];      // ACOL store coords: 4-row m group, k-pair
    int rM[4], rK[4];       // row-major store coords
    if (ACOL) {
        // tile viewed as 16 k-pairs x 32 m-quads; thread task: 2 rows x float4
        #pragma unroll
        for (int it = 0; it < 2; ++it) {
            const int i  = tid + it * 256;       // 0..511
            const int kp = i >> 5;               // k-pair index [0,16)
            const int m4 = (i & 31) << 2;
            aKp[it] = kp; aM[it] = m4;
            aP[2 * it]     = Ab + (long long)(2 * kp) * lda + mBase + m4;
            aP[2 * it + 1] = aP[2 * it] + lda;
        }
    } else {
        // tile viewed as 128 rows x 8 k-quads; thread task: one float4
        #pragma unroll
        for (int it = 0; it < 4; ++it) {
            const int i  = tid + it * 256;       // 0..1023
            const int m  = i >> 3;
            const int k4 = (i & 7) << 2;
            rM[it] = m; rK[it] = k4;
            aP[it] = Ab + (long long)(mBase + m) * lda + k4;
        }
    }
    const long long aStep = ACOL ? (long long)BK * lda : (long long)BK;

    constexpr int BTASK = (BK / 2) * (BN / 4);   // 256 (NT=4) or 128 (NT=2)
    const bool bAct  = (BTASK == 256) || (tid < BTASK);
    const int  btid  = tid & (BTASK - 1);
    const int  bKp   = btid / (BN / 4);          // k-pair
    const int  bN4   = (btid % (BN / 4)) << 2;   // n quad
    const float* bP0 = B + (long long)(2 * bKp) * ldb + nBase + bN4;
    const float* bP1 = bP0 + ldb;
    const long long bStep = (long long)BK * ldb;

    v8f acc[NT] = {};

    // ---- prime the register pipeline (all loads issue back-to-back) ----
    float4 aR[4], bR[2];
    #pragma unroll
    for (int j = 0; j < 4; ++j)
        aR[j] = *reinterpret_cast<const float4*>(aP[j]);
    if (bAct) {
        bR[0] = *reinterpret_cast<const float4*>(bP0);
        bR[1] = *reinterpret_cast<const float4*>(bP1);
    }

    const int am = wave * 16 + l16;              // A fragment LDS row

    #pragma unroll
    for (int t = 0; t < KT; ++t) {
        const int cur = t & 1;

        // ---- store staged registers to LDS as packed f16 ----
        if (ACOL) {
            #pragma unroll
            for (int it = 0; it < 2; ++it) {
                const float* f0 = reinterpret_cast<const float*>(&aR[2 * it]);
                const float* f1 = reinterpret_cast<const float*>(&aR[2 * it + 1]);
                #pragma unroll
                for (int j = 0; j < 4; ++j) {
                    PackU p;
                    p.h[0] = (_Float16)f0[j];    // k = 2*kp
                    p.h[1] = (_Float16)f1[j];    // k = 2*kp+1
                    *reinterpret_cast<unsigned*>(&Al[cur][aM[it] + j][2 * aKp[it]]) = p.u;
                }
            }
        } else {
            #pragma unroll
            for (int it = 0; it < 4; ++it) {
                Pack2U p;
                p.h[0] = (_Float16)aR[it].x;
                p.h[1] = (_Float16)aR[it].y;
                p.h[2] = (_Float16)aR[it].z;
                p.h[3] = (_Float16)aR[it].w;
                *reinterpret_cast<uint2*>(&Al[cur][rM[it]][rK[it]]) = p.u2;
            }
        }
        if (bAct) {
            const float* f0 = reinterpret_cast<const float*>(&bR[0]);
            const float* f1 = reinterpret_cast<const float*>(&bR[1]);
            #pragma unroll
            for (int j = 0; j < 4; ++j) {
                PackU p;
                p.h[0] = (_Float16)f0[j];
                p.h[1] = (_Float16)f1[j];
                *reinterpret_cast<unsigned*>(&Bl[cur][bN4 + j][2 * bKp]) = p.u;
            }
        }

        // ---- issue next tile's global loads (overlap with WMMA below) ----
        if (t + 1 < KT) {
            #pragma unroll
            for (int j = 0; j < 4; ++j) {
                aP[j] += aStep;
                aR[j] = *reinterpret_cast<const float4*>(aP[j]);
            }
            if (bAct) {
                bP0 += bStep; bP1 += bStep;
                bR[0] = *reinterpret_cast<const float4*>(bP0);
                bR[1] = *reinterpret_cast<const float4*>(bP1);
            }
        }

        __syncthreads();   // stores to buf[cur] visible; reads of buf[cur^1]
                           // from tile t-1 completed (compiler waits dscnt)

        // A fragment: 16-bit A 16x32 layout (ISA 7.12.2):
        //   lane m=l16; lanes 0-15: K 0..7 & 16..23, lanes 16-31: K 8..15 & 24..31
        FragH af;
        af.u[0] = *reinterpret_cast<const uint4*>(&Al[cur][am][lhi * 8]);
        af.u[1] = *reinterpret_cast<const uint4*>(&Al[cur][am][16 + lhi * 8]);

        #pragma unroll
        for (int nt = 0; nt < NT; ++nt) {
            // B fragment: lane n=l16; lanes 0-15: K 0..7 & 16..23, lanes 16-31: K 8..15 & 24..31
            FragH bf;
            const int bn = nt * 16 + l16;
            bf.u[0] = *reinterpret_cast<const uint4*>(&Bl[cur][bn][lhi * 8]);
            bf.u[1] = *reinterpret_cast<const uint4*>(&Bl[cur][bn][16 + lhi * 8]);
            acc[nt] = __builtin_amdgcn_wmma_f32_16x16x32_f16(
                false, af.v, false, bf.v, (short)0, acc[nt], false, false);
        }
    }

    // ---- epilogue: bias (+ camera-ray rank-3 term), store fp32 ----
    float camv[8][3];
    if (CAM) {
        const float* cb = cam + batch * camBatch;
        #pragma unroll
        for (int r = 0; r < 8; ++r) {
            const long long row = (long long)mBase + wave * 16 + r + lhi * 8;
            camv[r][0] = cb[row * 3 + 0];
            camv[r][1] = cb[row * 3 + 1];
            camv[r][2] = cb[row * 3 + 2];
        }
    }
    #pragma unroll
    for (int nt = 0; nt < NT; ++nt) {
        const int col = nBase + nt * 16 + l16;
        const float bv = bias[col];
        float w0 = 0.f, w1 = 0.f, w2 = 0.f;
        if (CAM) {
            w0 = Wcam[0 * ldb + col];
            w1 = Wcam[1 * ldb + col];
            w2 = Wcam[2 * ldb + col];
        }
        #pragma unroll
        for (int r = 0; r < 8; ++r) {
            const long long row = (long long)mBase + wave * 16 + r + lhi * 8;
            float v = acc[nt][r] + bv;
            if (CAM) v += camv[r][0] * w0 + camv[r][1] * w1 + camv[r][2] * w2;
            Cb[row * ldc + col] = v;
        }
    }
}

// ---------------------------------------------------------------------------
// feat = bilinear(src_hw, ref) + query.  One block per (v,q), thread = channel.
// src0 is L2-resident (157MB < 192MB L2) so the strided gather hits L2.
// ---------------------------------------------------------------------------
__global__ __launch_bounds__(256)
void feat_kernel(const float* __restrict__ query, const float* __restrict__ refp,
                 const float* __restrict__ src0, float* __restrict__ feat)
{
    const int vq = blockIdx.x;          // v*LQ_ + q
    const int v  = vq / LQ_;
    const int c  = threadIdx.x;

    const float rx = refp[vq * 2 + 0];
    const float ry = refp[vq * 2 + 1];
    const float gx = fminf(fmaxf(rx * 2.f - 1.f, -1.1f), 1.1f);
    const float gy = fminf(fmaxf(ry * 2.f - 1.f, -1.1f), 1.1f);
    const float x  = (gx + 1.f) * (W_ * 0.5f) - 0.5f;
    const float y  = (gy + 1.f) * (H_ * 0.5f) - 0.5f;
    const float x0f = floorf(x), y0f = floorf(y);
    const float dx = x - x0f, dy = y - y0f;
    const int x0 = (int)x0f, y0 = (int)y0f;

    const float* plane = src0 + ((long long)v * C_ + c) * HW_;
    auto g = [&](int yi, int xi) -> float {
        if (yi < 0 || yi >= H_ || xi < 0 || xi >= W_) return 0.f;
        return plane[yi * W_ + xi];
    };
    const float v00 = g(y0, x0),     v01 = g(y0, x0 + 1);
    const float v10 = g(y0 + 1, x0), v11 = g(y0 + 1, x0 + 1);
    const float s = v00 * (1.f - dx) * (1.f - dy) + v01 * dx * (1.f - dy)
                  + v10 * (1.f - dx) * dy         + v11 * dx * dy;
    feat[(long long)vq * C_ + c] = s + query[(long long)vq * C_ + c];
}

// ---------------------------------------------------------------------------
// Deformable sampling: one wave per (v,q,head); lane = dh channel so each
// corner read is a coalesced 128B burst from `value`. 4-way softmax inline.
// ---------------------------------------------------------------------------
__global__ __launch_bounds__(256)
void sample_kernel(const float* __restrict__ refp, const float* __restrict__ so,
                   const float* __restrict__ awl, const float* __restrict__ value,
                   float* __restrict__ attn)
{
    const int wid  = blockIdx.x * 8 + (threadIdx.x >> 5);   // (v,q,h) index
    const int lane = threadIdx.x & 31;                      // d in [0,32)
    const int h    = wid & (NH_ - 1);
    const int vq   = wid >> 3;
    const int v    = vq / LQ_;

    const float rx = refp[vq * 2 + 0] * (float)W_ - 0.5f;
    const float ry = refp[vq * 2 + 1] * (float)H_ - 0.5f;

    // softmax over the 4 point logits of this head
    const float* al = awl + (long long)vq * (NH_ * P_) + h * P_;
    const float a0 = al[0], a1 = al[1], a2 = al[2], a3 = al[3];
    const float mx = fmaxf(fmaxf(a0, a1), fmaxf(a2, a3));
    const float e0 = __expf(a0 - mx), e1 = __expf(a1 - mx);
    const float e2 = __expf(a2 - mx), e3 = __expf(a3 - mx);
    const float inv = 1.f / (e0 + e1 + e2 + e3);
    const float wp[4] = { e0 * inv, e1 * inv, e2 * inv, e3 * inv };

    const float* sop   = so + (long long)vq * (NH_ * P_ * 2) + h * (P_ * 2);
    const float* vbase = value + (long long)v * HW_ * C_ + h * DH_ + lane;

    float acc = 0.f;
    #pragma unroll
    for (int p = 0; p < P_; ++p) {
        const float x = rx + sop[p * 2 + 0];   // (ref + so/W)*W - 0.5
        const float y = ry + sop[p * 2 + 1];
        const float x0f = floorf(x), y0f = floorf(y);
        const float dx = x - x0f, dy = y - y0f;
        const int x0 = (int)x0f, y0 = (int)y0f;
        auto ld = [&](int yi, int xi) -> float {
            if (yi < 0 || yi >= H_ || xi < 0 || xi >= W_) return 0.f;
            return vbase[((long long)yi * W_ + xi) * C_];
        };
        const float v00 = ld(y0, x0),     v01 = ld(y0, x0 + 1);
        const float v10 = ld(y0 + 1, x0), v11 = ld(y0 + 1, x0 + 1);
        const float bil = v00 * (1.f - dx) * (1.f - dy) + v01 * dx * (1.f - dy)
                        + v10 * (1.f - dx) * dy         + v11 * dx * dy;
        acc += wp[p] * bil;
    }
    attn[(long long)vq * C_ + h * DH_ + lane] = acc;
}

// ---------------------------------------------------------------------------
extern "C" void kernel_launch(void* const* d_in, const int* in_sizes, int n_in,
                              void* d_out, int out_size, void* d_ws, size_t ws_size,
                              hipStream_t stream)
{
    (void)in_sizes; (void)n_in; (void)out_size; (void)ws_size;

    const float* query = (const float*)d_in[0];
    const float* refp  = (const float*)d_in[1];
    const float* src0  = (const float*)d_in[2];
    const float* cam   = (const float*)d_in[3];
    // d_in[4] / d_in[5]: spatial shapes / level starts -- compile-time constants
    const float* W_ray = (const float*)d_in[6];
    const float* b_ray = (const float*)d_in[7];
    const float* W_so  = (const float*)d_in[8];
    const float* b_so  = (const float*)d_in[9];
    const float* W_aw  = (const float*)d_in[10];
    const float* b_aw  = (const float*)d_in[11];
    const float* W_out = (const float*)d_in[12];
    const float* b_out = (const float*)d_in[13];
    float* out = (float*)d_out;

    // workspace carving (floats): value | feat | so | aw | attn  (~207MB)
    float* value = (float*)d_ws;
    float* feat  = value + (size_t)V_ * HW_ * C_;
    float* sob   = feat  + (size_t)V_ * LQ_ * C_;
    float* awb   = sob   + (size_t)V_ * LQ_ * (NH_ * P_ * 2);
    float* attn  = awb   + (size_t)V_ * LQ_ * (NH_ * P_);

    // 1) value[v,hw,:] = src0^T @ W_ray[:256] + cam @ W_ray[256:259] + b_ray
    //    A col-major (lda=HW), batched over views.
    gemm_k256<4, true, true><<<dim3(HW_ / 128, C_ / 64, V_), 256, 0, stream>>>(
        src0, (long long)C_ * HW_, HW_,
        W_ray, C_, b_ray,
        cam, (long long)HW_ * 3, W_ray + 256 * C_,
        value, (long long)HW_ * C_, C_);

    // 2) feat = bilinear(src0, ref) + query
    feat_kernel<<<dim3(V_ * LQ_), 256, 0, stream>>>(query, refp, src0, feat);

    // 3) sampling offsets & attention logits (K=256 GEMMs over feat)
    gemm_k256<4, false, false><<<dim3((V_ * LQ_) / 128, 1, 1), 256, 0, stream>>>(
        feat, 0, C_, W_so, 64, b_so, nullptr, 0, nullptr, sob, 0, 64);
    gemm_k256<2, false, false><<<dim3((V_ * LQ_) / 128, 1, 1), 256, 0, stream>>>(
        feat, 0, C_, W_aw, 32, b_aw, nullptr, 0, nullptr, awb, 0, 32);

    // 4) deformable bilinear sampling + softmax-weighted head reduction
    sample_kernel<<<dim3((V_ * LQ_ * NH_) / 8), 256, 0, stream>>>(
        refp, sob, awb, value, attn);

    // 5) output projection: out = attn @ W_out + b_out
    gemm_k256<4, false, false><<<dim3((V_ * LQ_) / 128, C_ / 64, 1), 256, 0, stream>>>(
        attn, 0, C_, W_out, C_, b_out, nullptr, 0, nullptr, out, 0, C_);
}
